// GRU_26740466385736
// MI455X (gfx1250) — matrix operand order repaired and checked by reading, not compiled, not run on hardware
//
#include <hip/hip_runtime.h>

// ---------------------------------------------------------------------------
// GRU forward on gfx1250 (CDNA5): bf16 WMMA 16x16x32, f32 accumulate.
//   T=512, B=64, INPUT=512, HIDDEN=1024
// Strategy:
//   - precompute input projections xr/xz/xn with one big WMMA GEMM (parallel over T)
//   - 512 sequential small per-step kernels for the recurrent part; hidden
//     weights pre-packed into WMMA-fragment tiles (6MB bf16, L2-resident)
//   - per-step h tile staged into LDS with GLOBAL_LOAD_ASYNC_TO_LDS (ASYNCcnt)
// ---------------------------------------------------------------------------

typedef __bf16 bf16;
typedef __attribute__((ext_vector_type(8)))  bf16  v8bf;
typedef __attribute__((ext_vector_type(16))) bf16  v16bf;
typedef __attribute__((ext_vector_type(8)))  float v8f;
typedef __attribute__((ext_vector_type(4)))  int   v4i;

#define T_STEPS 512
#define BATCH   64
#define ISZ     512
#define HSZ     1024
#define TB      (T_STEPS * BATCH)          // 32768 rows for the input GEMM

#define AS_GLOBAL __attribute__((address_space(1)))
#define AS_LOCAL  __attribute__((address_space(3)))

__device__ __forceinline__ v16bf cat8(v8bf lo, v8bf hi) {
  return __builtin_shufflevector(lo, hi, 0,1,2,3,4,5,6,7,8,9,10,11,12,13,14,15);
}

__device__ __forceinline__ v8f wmma_bf16(v16bf a, v16bf b, v8f c) {
  // D = A(16x32) * B(32x16) + C, f32 accumulate
  return __builtin_amdgcn_wmma_f32_16x16x32_bf16(
      /*neg_a=*/false, a, /*neg_b=*/false, b,
      /*c_mod=*/(short)0, c, /*reuse_a=*/false, /*reuse_b=*/false);
}

__device__ __forceinline__ float sigmoidf_(float x) {
  return 1.0f / (1.0f + __expf(-x));
}
__device__ __forceinline__ float tanhf_(float x) {
  float e = __expf(-2.0f * fabsf(x));
  float r = (1.0f - e) / (1.0f + e);
  return copysignf(r, x);
}

// ---------------------------------------------------------------------------
// f32 -> bf16 elementwise
// ---------------------------------------------------------------------------
__global__ void f32_to_bf16_kernel(const float* __restrict__ src,
                                   bf16* __restrict__ dst, int n) {
  int i = blockIdx.x * blockDim.x + threadIdx.x;
  if (i < n) dst[i] = (bf16)src[i];
}

// ---------------------------------------------------------------------------
// Pack W[K][N] (row-major f32) into WMMA B-fragment tiles:
//   layout [N/16][K/32][512] bf16; within a 512-elem tile:
//     lane = (n%16) + 16*h, element = lane*16 + s, k = 8h + (s&7) + 16*(s>>3)
//   (B 32x16: lanes 0-15 hold K=0..7,16..23, lanes 16-31 hold K=8..15,24..31; N = lane%16)
// ---------------------------------------------------------------------------
__global__ void pack_w_kernel(const float* __restrict__ W,
                              bf16* __restrict__ dst, int K, int N) {
  int e = blockIdx.x * blockDim.x + threadIdx.x;
  if (e >= K * N) return;
  int within  = e & 511;
  int tileIdx = e >> 9;
  int lane    = within >> 4;
  int slot    = within & 15;          // == (k & 7) + 8*((k >> 4) & 1)
  int kcCount = K >> 5;
  int kc      = tileIdx % kcCount;
  int ntile   = tileIdx / kcCount;
  int k = kc * 32 + (lane >> 4) * 8 + (slot & 7) + (slot >> 3) * 16;
  int n = ntile * 16 + (lane & 15);
  dst[e] = (bf16)W[(size_t)k * N + n];
}

// ---------------------------------------------------------------------------
// Input projections: xrzn[g][row][h] = x[row] @ W_i(g) + b_i(g),  row = t*B+b
// One wave per 16x16 output tile per gate. K = 512 -> 16 WMMAs per wave.
// ---------------------------------------------------------------------------
__global__ __launch_bounds__(256)
void gru_input_proj_kernel(const bf16* __restrict__ xb,    // [TB][ISZ] bf16
                           const bf16* __restrict__ Wip,   // packed, gate stride ISZ*HSZ
                           const float* __restrict__ b_ir,
                           const float* __restrict__ b_iz,
                           const float* __restrict__ b_in,
                           bf16* __restrict__ xrzn)        // [3][TB][HSZ]
{
  int wave = threadIdx.x >> 5;
  int lane = threadIdx.x & 31;
  int wid  = blockIdx.x * 8 + wave;        // 0 .. 2048*64*3-1
  int ntile = wid % 64;
  int gate  = (wid / 64) % 3;
  int mtile = wid / 192;

  const bf16* Wt = Wip + ((size_t)(gate * 64 + ntile) * 16) * 512;

  int m  = lane & 15;
  int kh = lane >> 4;
  const bf16* arow = xb + (size_t)(mtile * 16 + m) * ISZ + kh * 8;

  v8f acc = {};
#pragma unroll 4
  for (int kc = 0; kc < 16; ++kc) {
    const v8bf* p = (const v8bf*)(arow + kc * 32);
    v16bf a = cat8(p[0], p[2]);                      // K groups {0..7,16..23}(+kh*8)
    const v8bf* q = (const v8bf*)(Wt + (size_t)kc * 512 + lane * 16);
    v16bf b = cat8(q[0], q[1]);
    acc = wmma_bf16(a, b, acc);
  }

  const float* bias = (gate == 0) ? b_ir : ((gate == 1) ? b_iz : b_in);
  int   ncol = ntile * 16 + (lane & 15);
  float bv   = bias[ncol];
  int   mh   = lane >> 4;
  size_t gbase = (size_t)gate * TB;
#pragma unroll
  for (int r = 0; r < 8; ++r) {
    int row = mtile * 16 + r + 8 * mh;
    xrzn[(gbase + row) * HSZ + ncol] = (bf16)(acc[r] + bv);
  }
}

// ---------------------------------------------------------------------------
// One GRU timestep. 32 blocks x 256 threads. Each wave: one (16 batch x 16
// hidden) tile, all three gates (3 x 32 WMMAs over K=1024). h staged in LDS
// via the async global->LDS datapath (no VGPR round-trip, ASYNCcnt tracked).
// ---------------------------------------------------------------------------
__global__ __launch_bounds__(256)
void gru_cell_kernel(const bf16* __restrict__ h_in,   // [B][H] bf16
                     bf16* __restrict__ h_out,        // [B][H] bf16
                     const bf16* __restrict__ Whp,    // packed, gate stride H*H
                     const float* __restrict__ b_hr,
                     const float* __restrict__ b_hz,
                     const float* __restrict__ b_hn,
                     const bf16* __restrict__ xrzn,   // [3][TB][H]
                     float* __restrict__ out_t,       // d_out + t*B*H
                     int t)
{
  __shared__ __align__(16) bf16 hsm[16 * HSZ];        // 32 KB: this block's 16 h-rows

  int tid  = threadIdx.x;
  int wave = tid >> 5;
  int lane = tid & 31;
  int mtile = blockIdx.x >> 3;                        // 0..3
  int ntile = (blockIdx.x & 7) * 8 + wave;            // 0..63

  // stage 16 contiguous h rows (32 KB) into LDS: 2048 x b128, 8 per thread
  {
    const bf16* src = h_in + (size_t)mtile * 16 * HSZ;
#if __has_builtin(__builtin_amdgcn_global_load_async_to_lds_b128)
#pragma unroll
    for (int j = 0; j < 8; ++j) {
      int i = tid + j * 256;                          // b128 index 0..2047
      __builtin_amdgcn_global_load_async_to_lds_b128(
          (AS_GLOBAL v4i*)(src + i * 8),              // global src (16 B)
          (AS_LOCAL  v4i*)(hsm + i * 8),              // LDS dst   (16 B)
          /*offset=*/0, /*cpol=*/0);
    }
    asm volatile("s_wait_asynccnt 0" ::: "memory");
#else
    const uint4* s4 = (const uint4*)src;
    uint4* d4 = (uint4*)hsm;
#pragma unroll
    for (int j = 0; j < 8; ++j) {
      int i = tid + j * 256;
      d4[i] = s4[i];
    }
#endif
  }
  __syncthreads();

  int m  = lane & 15;
  int kh = lane >> 4;
  const bf16* hrow = hsm + m * HSZ + kh * 8;
  const bf16* W0 = Whp + ((size_t)(  0 + ntile) * 32) * 512 + lane * 16;
  const bf16* W1 = Whp + ((size_t)( 64 + ntile) * 32) * 512 + lane * 16;
  const bf16* W2 = Whp + ((size_t)(128 + ntile) * 32) * 512 + lane * 16;

  v8f ar = {}, az = {}, an = {};
#pragma unroll 4
  for (int kc = 0; kc < 32; ++kc) {
    const v8bf* p = (const v8bf*)(hrow + kc * 32);    // ds_load_b128 x2
    v16bf a = cat8(p[0], p[2]);
    { const v8bf* q = (const v8bf*)(W0 + (size_t)kc * 512);
      ar = wmma_bf16(a, cat8(q[0], q[1]), ar); }
    { const v8bf* q = (const v8bf*)(W1 + (size_t)kc * 512);
      az = wmma_bf16(a, cat8(q[0], q[1]), az); }
    { const v8bf* q = (const v8bf*)(W2 + (size_t)kc * 512);
      an = wmma_bf16(a, cat8(q[0], q[1]), an); }
  }

  int   ncol = ntile * 16 + (lane & 15);
  float br = b_hr[ncol], bz = b_hz[ncol], bn = b_hn[ncol];
  int   mh = lane >> 4;
  const size_t GSTRIDE = (size_t)TB * HSZ;

#pragma unroll
  for (int r = 0; r < 8; ++r) {
    int   ml   = r + 8 * mh;                 // local batch row 0..15
    int   brow = mtile * 16 + ml;            // global batch row
    float hold = (float)hsm[ml * HSZ + ncol];
    size_t xoff = (size_t)(t * BATCH + brow) * HSZ + ncol;
    float xr = (float)xrzn[xoff];
    float xz = (float)xrzn[GSTRIDE + xoff];
    float xn = (float)xrzn[2 * GSTRIDE + xoff];
    float rt = sigmoidf_(ar[r] + br + xr);
    float zt = sigmoidf_(az[r] + bz + xz);
    float nt = tanhf_(xn + rt * (an[r] + bn));
    float ht = (1.0f - zt) * nt + zt * hold;
    out_t[(size_t)brow * HSZ + ncol] = ht;
    h_out[(size_t)brow * HSZ + ncol] = (bf16)ht;
  }
}

// ---------------------------------------------------------------------------
// Launch
// ---------------------------------------------------------------------------
extern "C" void kernel_launch(void* const* d_in, const int* in_sizes, int n_in,
                              void* d_out, int out_size, void* d_ws, size_t ws_size,
                              hipStream_t stream) {
  const float* x    = (const float*)d_in[0];
  const float* h0   = (const float*)d_in[1];
  const float* W_ir = (const float*)d_in[2];
  const float* b_ir = (const float*)d_in[3];
  const float* W_hr = (const float*)d_in[4];
  const float* b_hr = (const float*)d_in[5];
  const float* W_iz = (const float*)d_in[6];
  const float* b_iz = (const float*)d_in[7];
  const float* W_hz = (const float*)d_in[8];
  const float* b_hz = (const float*)d_in[9];
  const float* W_in = (const float*)d_in[10];
  const float* b_in = (const float*)d_in[11];
  const float* W_hn = (const float*)d_in[12];
  const float* b_hn = (const float*)d_in[13];
  float* out = (float*)d_out;

  // workspace layout (bytes), all offsets 256-aligned
  char* ws = (char*)d_ws;
  const size_t OFF_XB   = 0;                                   // TB*ISZ bf16   = 32 MB
  const size_t OFF_XRZN = OFF_XB   + (size_t)TB * ISZ * 2;     // 3*TB*HSZ bf16 = 192 MB
  const size_t OFF_WHP  = OFF_XRZN + (size_t)3 * TB * HSZ * 2; // 3*H*H bf16    = 6 MB
  const size_t OFF_WIP  = OFF_WHP  + (size_t)3 * HSZ * HSZ * 2;// 3*I*H bf16    = 3 MB
  const size_t OFF_HB   = OFF_WIP  + (size_t)3 * ISZ * HSZ * 2;// 2*B*H bf16

  bf16* xb    = (bf16*)(ws + OFF_XB);
  bf16* xrzn  = (bf16*)(ws + OFF_XRZN);
  bf16* Whp   = (bf16*)(ws + OFF_WHP);
  bf16* Wip   = (bf16*)(ws + OFF_WIP);
  bf16* hbuf0 = (bf16*)(ws + OFF_HB);
  bf16* hbuf1 = hbuf0 + (size_t)BATCH * HSZ;

  // 1) convert x and h0 to bf16
  {
    int n = TB * ISZ;
    f32_to_bf16_kernel<<<(n + 255) / 256, 256, 0, stream>>>(x, xb, n);
  }
  {
    int n = BATCH * HSZ;
    f32_to_bf16_kernel<<<(n + 255) / 256, 256, 0, stream>>>(h0, hbuf0, n);
  }

  // 2) pack all six weight matrices into WMMA B-fragment tile order
  {
    int nI = ISZ * HSZ, nH = HSZ * HSZ;
    pack_w_kernel<<<(nI + 255) / 256, 256, 0, stream>>>(W_ir, Wip + 0 * (size_t)nI, ISZ, HSZ);
    pack_w_kernel<<<(nI + 255) / 256, 256, 0, stream>>>(W_iz, Wip + 1 * (size_t)nI, ISZ, HSZ);
    pack_w_kernel<<<(nI + 255) / 256, 256, 0, stream>>>(W_in, Wip + 2 * (size_t)nI, ISZ, HSZ);
    pack_w_kernel<<<(nH + 255) / 256, 256, 0, stream>>>(W_hr, Whp + 0 * (size_t)nH, HSZ, HSZ);
    pack_w_kernel<<<(nH + 255) / 256, 256, 0, stream>>>(W_hz, Whp + 1 * (size_t)nH, HSZ, HSZ);
    pack_w_kernel<<<(nH + 255) / 256, 256, 0, stream>>>(W_hn, Whp + 2 * (size_t)nH, HSZ, HSZ);
  }

  // 3) input projections: (TB/16) * (HSZ/16) * 3 gates waves, 8 waves/block
  {
    int totalWaves = (TB / 16) * (HSZ / 16) * 3;   // 393216
    gru_input_proj_kernel<<<totalWaves / 8, 256, 0, stream>>>(
        xb, Wip, b_ir, b_iz, b_in, xrzn);
  }

  // 4) 512 sequential GRU cell steps (ping-pong bf16 hidden state)
  for (int t = 0; t < T_STEPS; ++t) {
    const bf16* hin = (t & 1) ? hbuf1 : hbuf0;
    bf16*      hout = (t & 1) ? hbuf0 : hbuf1;
    gru_cell_kernel<<<32, 256, 0, stream>>>(
        hin, hout, Whp, b_hr, b_hz, b_hn, xrzn,
        out + (size_t)t * BATCH * HSZ, t);
  }
}
